// BaseModel_58978490909176
// MI455X (gfx1250) — compile-verified, matrix operand order checked by reading, not run on hardware
//
#include <hip/hip_runtime.h>
#include <hip/hip_bf16.h>

#define HDIM 1900
#define HP   1920          // padded H (multiple of 128)
#define NB   256
#define T_EPI 25
#define T_TOT 153
#define ICN  3800
#define OCN  380
#define PADTOK 26

typedef __attribute__((ext_vector_type(16))) __bf16 v16bf;
typedef __attribute__((ext_vector_type(8)))  float  v8f;

// ---- CDNA5 async global->LDS copy (ASYNCcnt-tracked, bypasses VGPRs) --------
__device__ __forceinline__ void async_b128(unsigned lds_off, const void* g) {
  asm volatile("global_load_async_to_lds_b128 %0, %1, off" :: "v"(lds_off), "v"(g) : "memory");
}
__device__ __forceinline__ void wait_async0() {
  asm volatile("s_wait_asynccnt 0x0" ::: "memory");
}
// low 32 bits of a generic pointer to __shared__ = LDS byte offset
__device__ __forceinline__ unsigned lds_off_u32(const void* p) {
  return (unsigned)(unsigned long long)p;
}

// ---------------------------------------------------------------- setup kernels

__global__ void colnorm_kernel(const float* __restrict__ w, const float* __restrict__ g,
                               float* __restrict__ out, int R, int C) {
  int c = blockIdx.x * blockDim.x + threadIdx.x;
  if (c >= C) return;
  float s = 0.f;
  for (int r = 0; r < R; ++r) { float v = w[(size_t)r * C + c]; s += v * v; }
  out[c] = g[c] * rsqrtf(fmaxf(s, 1e-12f));
}

// whT[g][n][k] = bf16(whn[k][g*H+n]), zero-padded to HPxHP
__global__ void conv_whT_kernel(const float* __restrict__ wh, const float* __restrict__ scale,
                                __hip_bfloat16* __restrict__ whT) {
  int k = blockIdx.x * blockDim.x + threadIdx.x;
  int n = blockIdx.y;
  int g = blockIdx.z;
  if (k >= HP) return;
  float v = 0.f;
  if (k < HDIM && n < HDIM) {
    int col = g * HDIM + n;
    v = wh[(size_t)k * (4 * HDIM) + col] * scale[col];
  }
  whT[((size_t)(g * HP + n)) * HP + k] = __float2bfloat16(v);
}

// wmhT[n][k] = bf16(wmhn[k][n]), zero-padded
__global__ void conv_wmhT_kernel(const float* __restrict__ wmh, const float* __restrict__ scale,
                                 __hip_bfloat16* __restrict__ wmhT) {
  int k = blockIdx.x * blockDim.x + threadIdx.x;
  int n = blockIdx.y;
  if (k >= HP) return;
  float v = 0.f;
  if (k < HDIM && n < HDIM) v = wmh[(size_t)k * HDIM + n] * scale[n];
  wmhT[(size_t)n * HP + k] = __float2bfloat16(v);
}

// proj_mx[tok][n] = (embed[tok] @ wmxn)[n];  proj_x[g][tok][n] = (embed[tok] @ wxn)[g*H+n] + b[g*H+n]
__global__ void proj_kernel(const float* __restrict__ embed, const float* __restrict__ wx,
                            const float* __restrict__ wmx, const float* __restrict__ bias,
                            const float* __restrict__ scale_x, const float* __restrict__ scale_mx,
                            float* __restrict__ proj_x, float* __restrict__ proj_mx) {
  int n = blockIdx.x * blockDim.x + threadIdx.x;
  int tok = blockIdx.y;
  if (n >= HP) return;
  float pm = 0.f;
  if (n < HDIM) {
    float s = 0.f;
    for (int e = 0; e < 10; ++e) s += embed[tok * 10 + e] * wmx[e * HDIM + n];
    pm = s * scale_mx[n];
  }
  proj_mx[tok * HP + n] = pm;
  for (int g = 0; g < 4; ++g) {
    float p = 0.f;
    if (n < HDIM) {
      int col = g * HDIM + n;
      float s = 0.f;
      for (int e = 0; e < 10; ++e) s += embed[tok * 10 + e] * wx[(size_t)e * (4 * HDIM) + col];
      p = s * scale_x[col] + bias[col];
    }
    proj_x[(size_t)(g * 27 + tok) * HP + n] = p;
  }
}

// select indices: idx[b] = ti (total sequence), idx[NB+b] = ei (epitope sequence)
__global__ void len_kernel(const int* __restrict__ epi, const int* __restrict__ left,
                           const int* __restrict__ right, int* __restrict__ idx) {
  int b = blockIdx.x * blockDim.x + threadIdx.x;
  if (b >= NB) return;
  int el = 0; for (int t = 0; t < T_EPI; ++t) el += (epi[b * T_EPI + t] != PADTOK);
  int ll = 0; for (int t = 0; t < 64; ++t)    ll += (left[b * 64 + t] != PADTOK);
  int rl = 0; for (int t = 0; t < 64; ++t)    rl += (right[b * 64 + t] != PADTOK);
  if (ll < 1) ll = 1;
  if (rl < 1) rl = 1;
  int ti = el + ll + rl - 1;
  ti = ti < 0 ? 0 : (ti > T_TOT - 1 ? T_TOT - 1 : ti);
  int ei = el - 1;
  ei = ei < 0 ? 0 : (ei > T_EPI - 1 ? T_EPI - 1 : ei);
  idx[b] = ti;
  idx[NB + b] = ei;
}

// ---------------------------------------------------------------- recurrence GEMMs

__device__ __forceinline__ int row_token(int row, int t,
                                         const int* __restrict__ tot_tok,
                                         const int* __restrict__ epi_tok) {
  int tok = (row < NB) ? tot_tok[row * T_TOT + t] : epi_tok[(row - NB) * T_EPI + t];
  return tok < 0 ? 0 : (tok > 26 ? 26 : tok);
}

// m = (h @ wmhn) * proj_mx[token] -> m_bf16.  WG tile 64x128, wave tile 32x32.
__global__ __launch_bounds__(256)
void gemm_m_kernel(const __hip_bfloat16* __restrict__ hbuf, const __hip_bfloat16* __restrict__ wmhT,
                   const float* __restrict__ proj_mx,
                   const int* __restrict__ tot_tok, const int* __restrict__ epi_tok,
                   int t, __hip_bfloat16* __restrict__ mbuf) {
  __shared__ unsigned short A_lds[2][64 * 40];
  __shared__ unsigned short B_lds[2][128 * 40];
  const int tid = threadIdx.x;
  const int lane = tid & 31;
  const int wid = tid >> 5;
  const int wave_m = wid >> 2, wave_n = wid & 3;
  const int row0 = blockIdx.y * 64;
  const int col0 = blockIdx.x * 128;

  // cooperative tile-load mapping: 16B per thread per 64x32 tile
  const int l_r = (tid * 8) >> 5;   // 0..63
  const int l_c = (tid * 8) & 31;   // 0,8,16,24
  const unsigned short* gA  = (const unsigned short*)hbuf + (size_t)(row0 + l_r) * HP + l_c;
  const unsigned short* gB0 = (const unsigned short*)wmhT + (size_t)(col0 + l_r) * HP + l_c;
  const unsigned short* gB1 = (const unsigned short*)wmhT + (size_t)(col0 + 64 + l_r) * HP + l_c;
  const unsigned ldsA = lds_off_u32(&A_lds[0][0]) + (unsigned)(l_r * 40 + l_c) * 2u;
  const unsigned ldsB = lds_off_u32(&B_lds[0][0]) + (unsigned)(l_r * 40 + l_c) * 2u;
  const unsigned aStr = 64u * 40u * 2u;
  const unsigned bStr = 128u * 40u * 2u;

  v8f acc[2][2] = {};
  const int kh = (lane >> 4) * 8;
  const int lm = lane & 15;

  // prologue: tile 0 -> buffer 0
  async_b128(ldsA, gA);
  async_b128(ldsB, gB0);
  async_b128(ldsB + aStr, gB1);

  int cur = 0;
  for (int k0 = 0; k0 < HP; k0 += 32) {
    wait_async0();        // my tile-k loads landed
    __syncthreads();      // everyone's landed; everyone done reading the other buffer
    if (k0 + 32 < HP) {
      unsigned nb = (unsigned)(cur ^ 1);
      async_b128(ldsA + nb * aStr, gA + k0 + 32);
      async_b128(ldsB + nb * bStr, gB0 + k0 + 32);
      async_b128(ldsB + nb * bStr + aStr, gB1 + k0 + 32);
    }
    union { v16bf v; uint4 u[2]; } af[2], bfv[2];
#pragma unroll
    for (int sm = 0; sm < 2; ++sm) {
      int ml = wave_m * 32 + sm * 16 + lm;
      af[sm].u[0] = *(const uint4*)&A_lds[cur][ml * 40 + kh];
      af[sm].u[1] = *(const uint4*)&A_lds[cur][ml * 40 + 16 + kh];
    }
#pragma unroll
    for (int sn = 0; sn < 2; ++sn) {
      int nl = wave_n * 32 + sn * 16 + lm;
      bfv[sn].u[0] = *(const uint4*)&B_lds[cur][nl * 40 + kh];
      bfv[sn].u[1] = *(const uint4*)&B_lds[cur][nl * 40 + 16 + kh];
    }
#pragma unroll
    for (int sm = 0; sm < 2; ++sm)
#pragma unroll
      for (int sn = 0; sn < 2; ++sn)
        acc[sm][sn] = __builtin_amdgcn_wmma_f32_16x16x32_bf16(
            false, af[sm].v, false, bfv[sn].v, (short)0, acc[sm][sn], false, false);
    cur ^= 1;
  }

  const int rtop = row0 + wave_m * 32 + (lane >> 4) * 8;
#pragma unroll
  for (int sm = 0; sm < 2; ++sm) {
#pragma unroll
    for (int r = 0; r < 8; ++r) {
      int row = rtop + sm * 16 + r;
      int tok = row_token(row, t, tot_tok, epi_tok);
#pragma unroll
      for (int sn = 0; sn < 2; ++sn) {
        int ncol = col0 + wave_n * 32 + sn * 16 + lm;
        float v = acc[sm][sn][r] * proj_mx[tok * HP + ncol];
        mbuf[(size_t)row * HP + ncol] = __float2bfloat16(v);
      }
    }
  }
}

// z[g] = m @ whn[:, gH:(g+1)H] + proj_x[g][token]; gates; update c, h.
// WG tile 64x64 per gate, wave tile 32x16 per gate (B fragment reused across 2 A fragments).
__global__ __launch_bounds__(256)
void gemm_gates_kernel(const __hip_bfloat16* __restrict__ mb, const __hip_bfloat16* __restrict__ whT,
                       const float* __restrict__ proj_x,
                       const int* __restrict__ tot_tok, const int* __restrict__ epi_tok,
                       const int* __restrict__ sel_idx, int t,
                       float* __restrict__ cbuf, __hip_bfloat16* __restrict__ hb,
                       float* __restrict__ selh) {
  __shared__ unsigned short A_lds[2][64 * 40];
  __shared__ unsigned short B_lds[2][4][64 * 40];
  const int tid = threadIdx.x;
  const int lane = tid & 31;
  const int wid = tid >> 5;
  const int wave_m = wid >> 2, wave_n = wid & 3;
  const int row0 = blockIdx.y * 64;
  const int col0 = blockIdx.x * 64;

  const int l_r = (tid * 8) >> 5;   // 0..63
  const int l_c = (tid * 8) & 31;
  const unsigned short* gA = (const unsigned short*)mb + (size_t)(row0 + l_r) * HP + l_c;
  const unsigned short* gB[4];
#pragma unroll
  for (int g = 0; g < 4; ++g)
    gB[g] = (const unsigned short*)whT + ((size_t)(g * HP + col0 + l_r)) * HP + l_c;
  const unsigned ldsA = lds_off_u32(&A_lds[0][0]) + (unsigned)(l_r * 40 + l_c) * 2u;
  const unsigned ldsB = lds_off_u32(&B_lds[0][0][0]) + (unsigned)(l_r * 40 + l_c) * 2u;
  const unsigned tStr = 64u * 40u * 2u;       // one 64x40 tile
  const unsigned bStr = 4u * tStr;            // one B buffer (4 gates)

  v8f acc[2][4] = {};
  const int kh = (lane >> 4) * 8;
  const int lm = lane & 15;

  // prologue
  async_b128(ldsA, gA);
#pragma unroll
  for (int g = 0; g < 4; ++g) async_b128(ldsB + (unsigned)g * tStr, gB[g]);

  int cur = 0;
  for (int k0 = 0; k0 < HP; k0 += 32) {
    wait_async0();
    __syncthreads();
    if (k0 + 32 < HP) {
      unsigned nb = (unsigned)(cur ^ 1);
      async_b128(ldsA + nb * tStr, gA + k0 + 32);
#pragma unroll
      for (int g = 0; g < 4; ++g)
        async_b128(ldsB + nb * bStr + (unsigned)g * tStr, gB[g] + k0 + 32);
    }
    union { v16bf v; uint4 u[2]; } af[2], bfv;
#pragma unroll
    for (int sm = 0; sm < 2; ++sm) {
      int ml = wave_m * 32 + sm * 16 + lm;
      af[sm].u[0] = *(const uint4*)&A_lds[cur][ml * 40 + kh];
      af[sm].u[1] = *(const uint4*)&A_lds[cur][ml * 40 + 16 + kh];
    }
    int nl = wave_n * 16 + lm;
#pragma unroll
    for (int g = 0; g < 4; ++g) {
      bfv.u[0] = *(const uint4*)&B_lds[cur][g][nl * 40 + kh];
      bfv.u[1] = *(const uint4*)&B_lds[cur][g][nl * 40 + 16 + kh];
      acc[0][g] = __builtin_amdgcn_wmma_f32_16x16x32_bf16(
          false, af[0].v, false, bfv.v, (short)0, acc[0][g], false, false);
      acc[1][g] = __builtin_amdgcn_wmma_f32_16x16x32_bf16(
          false, af[1].v, false, bfv.v, (short)0, acc[1][g], false, false);
    }
    cur ^= 1;
  }

  const int ncol = col0 + wave_n * 16 + lm;
  const int rtop = row0 + wave_m * 32 + (lane >> 4) * 8;
#pragma unroll
  for (int sm = 0; sm < 2; ++sm) {
#pragma unroll
    for (int r = 0; r < 8; ++r) {
      int row = rtop + sm * 16 + r;
      int tok = row_token(row, t, tot_tok, epi_tok);
      const float* px = proj_x + (size_t)tok * HP + ncol;
      float zi = acc[sm][0][r] + px[0 * 27 * HP];
      float zf = acc[sm][1][r] + px[1 * 27 * HP];
      float zo = acc[sm][2][r] + px[2 * 27 * HP];
      float zu = acc[sm][3][r] + px[3 * 27 * HP];
      float si = 1.f / (1.f + __expf(-zi));
      float sf = 1.f / (1.f + __expf(-zf));
      float so = 1.f / (1.f + __expf(-zo));
      size_t off = (size_t)row * HP + ncol;
      float cn = sf * cbuf[off] + si * tanhf(zu);
      float hn = so * tanhf(cn);
      cbuf[off] = cn;
      hb[off] = __float2bfloat16(hn);
      if (sel_idx[row] == t) selh[off] = hn;
    }
  }
}

// ---------------------------------------------------------------- classifier

__global__ __launch_bounds__(384)
void fc1_kernel(const float* __restrict__ selh,
                const float* __restrict__ g1, const float* __restrict__ be1,
                const float* __restrict__ mu1, const float* __restrict__ vr1,
                const float* __restrict__ W1, const float* __restrict__ b1,
                float* __restrict__ x1) {
  __shared__ float X[ICN];
  int b = blockIdx.x;
  for (int i = threadIdx.x; i < ICN; i += blockDim.x) {
    float v = (i < HDIM) ? selh[(size_t)b * HP + i]
                         : selh[(size_t)(NB + b) * HP + (i - HDIM)];
    v = v >= 0.f ? v : 0.3f * v;
    v = (v - mu1[i]) * rsqrtf(vr1[i] + 1e-3f) * g1[i] + be1[i];
    X[i] = v;
  }
  __syncthreads();
  int j = threadIdx.x;
  if (j < OCN) {
    float s = b1[j];
    for (int i = 0; i < ICN; ++i) s += X[i] * W1[(size_t)i * OCN + j];
    x1[b * OCN + j] = s;
  }
}

__global__ __launch_bounds__(256)
void fc2_kernel(const float* __restrict__ x1,
                const float* __restrict__ g2, const float* __restrict__ be2,
                const float* __restrict__ mu2, const float* __restrict__ vr2,
                const float* __restrict__ W2, const float* __restrict__ b2,
                float* __restrict__ out) {
  __shared__ float red[256];
  int b = blockIdx.x;
  float s = 0.f;
  for (int j = threadIdx.x; j < OCN; j += 256) {
    float v = x1[b * OCN + j];
    v = v >= 0.f ? v : 0.3f * v;
    v = (v - mu2[j]) * rsqrtf(vr2[j] + 1e-3f) * g2[j] + be2[j];
    s += v * W2[j];
  }
  red[threadIdx.x] = s;
  __syncthreads();
  for (int off = 128; off > 0; off >>= 1) {
    if (threadIdx.x < off) red[threadIdx.x] += red[threadIdx.x + off];
    __syncthreads();
  }
  if (threadIdx.x == 0) out[b] = red[0] + b2[0];
}

// ---------------------------------------------------------------- host driver

extern "C" void kernel_launch(void* const* d_in, const int* in_sizes, int n_in,
                              void* d_out, int out_size, void* d_ws, size_t ws_size,
                              hipStream_t stream) {
  const int*   epi_x   = (const int*)d_in[0];
  const int*   left_x  = (const int*)d_in[1];
  const int*   right_x = (const int*)d_in[2];
  const int*   tot_x   = (const int*)d_in[3];
  const float* embed   = (const float*)d_in[4];
  const float* wx      = (const float*)d_in[5];
  const float* wh      = (const float*)d_in[6];
  const float* wmx     = (const float*)d_in[7];
  const float* wmh     = (const float*)d_in[8];
  const float* bias    = (const float*)d_in[9];
  const float* gx      = (const float*)d_in[10];
  const float* gh      = (const float*)d_in[11];
  const float* gmx     = (const float*)d_in[12];
  const float* gmh     = (const float*)d_in[13];
  const float* bn1g    = (const float*)d_in[14];
  const float* bn1b    = (const float*)d_in[15];
  const float* bn1m    = (const float*)d_in[16];
  const float* bn1v    = (const float*)d_in[17];
  const float* W1      = (const float*)d_in[18];
  const float* b1      = (const float*)d_in[19];
  const float* bn2g    = (const float*)d_in[20];
  const float* bn2b    = (const float*)d_in[21];
  const float* bn2m    = (const float*)d_in[22];
  const float* bn2v    = (const float*)d_in[23];
  const float* W2      = (const float*)d_in[24];
  const float* b2      = (const float*)d_in[25];
  float* out = (float*)d_out;

  char* p = (char*)d_ws;
  auto carve = [&](size_t bytes) {
    char* q = p;
    p += (bytes + 255) & ~(size_t)255;
    return (void*)q;
  };
  float* scale_x  = (float*)carve(4 * HDIM * sizeof(float));
  float* scale_h  = (float*)carve(4 * HDIM * sizeof(float));
  float* scale_mx = (float*)carve(HDIM * sizeof(float));
  float* scale_mh = (float*)carve(HDIM * sizeof(float));
  __hip_bfloat16* whT  = (__hip_bfloat16*)carve((size_t)4 * HP * HP * 2);
  __hip_bfloat16* wmhT = (__hip_bfloat16*)carve((size_t)HP * HP * 2);
  float* proj_x  = (float*)carve((size_t)4 * 27 * HP * sizeof(float));
  float* proj_mx = (float*)carve((size_t)27 * HP * sizeof(float));
  float* cbuf    = (float*)carve((size_t)2 * NB * HP * sizeof(float));
  float* selh    = (float*)carve((size_t)2 * NB * HP * sizeof(float));
  __hip_bfloat16* hb = (__hip_bfloat16*)carve((size_t)2 * NB * HP * 2);
  __hip_bfloat16* mb = (__hip_bfloat16*)carve((size_t)2 * NB * HP * 2);
  int*   selidx  = (int*)carve(2 * NB * sizeof(int));
  float* x1      = (float*)carve((size_t)NB * OCN * sizeof(float));

  // ---- setup: weight-norm scales, transposed bf16 weights, token projection tables
  colnorm_kernel<<<dim3((4 * HDIM + 255) / 256), 256, 0, stream>>>(wx, gx, scale_x, 10, 4 * HDIM);
  colnorm_kernel<<<dim3((4 * HDIM + 255) / 256), 256, 0, stream>>>(wh, gh, scale_h, HDIM, 4 * HDIM);
  colnorm_kernel<<<dim3((HDIM + 255) / 256), 256, 0, stream>>>(wmx, gmx, scale_mx, 10, HDIM);
  colnorm_kernel<<<dim3((HDIM + 255) / 256), 256, 0, stream>>>(wmh, gmh, scale_mh, HDIM, HDIM);
  conv_whT_kernel<<<dim3((HP + 255) / 256, HP, 4), 256, 0, stream>>>(wh, scale_h, whT);
  conv_wmhT_kernel<<<dim3((HP + 255) / 256, HP), 256, 0, stream>>>(wmh, scale_mh, wmhT);
  proj_kernel<<<dim3((HP + 255) / 256, 27), 256, 0, stream>>>(embed, wx, wmx, bias,
                                                              scale_x, scale_mx, proj_x, proj_mx);
  len_kernel<<<dim3(1), 256, 0, stream>>>(epi_x, left_x, right_x, selidx);

  // ---- zero the recurrence state (every launch; deterministic)
  hipMemsetAsync(cbuf, 0, (size_t)2 * NB * HP * sizeof(float), stream);
  hipMemsetAsync(selh, 0, (size_t)2 * NB * HP * sizeof(float), stream);
  hipMemsetAsync(hb, 0, (size_t)2 * NB * HP * 2, stream);
  hipMemsetAsync(mb, 0, (size_t)2 * NB * HP * 2, stream);

  // ---- recurrence: rows 0..255 = total sequence, rows 256..511 = epitope sequence
  for (int t = 0; t < T_TOT; ++t) {
    int Bt = (t < T_EPI) ? 2 * NB : NB;
    gemm_m_kernel<<<dim3(HP / 128, Bt / 64), 256, 0, stream>>>(hb, wmhT, proj_mx,
                                                               tot_x, epi_x, t, mb);
    gemm_gates_kernel<<<dim3(HP / 64, Bt / 64), 256, 0, stream>>>(mb, whT, proj_x,
                                                                  tot_x, epi_x, selidx, t,
                                                                  cbuf, hb, selh);
  }

  // ---- classifier
  fc1_kernel<<<dim3(NB), 384, 0, stream>>>(selh, bn1g, bn1b, bn1m, bn1v, W1, b1, x1);
  fc2_kernel<<<dim3(NB), 256, 0, stream>>>(x1, bn2g, bn2b, bn2m, bn2v, W2, b2, out);
}
